// Cosface_19344532701363
// MI455X (gfx1250) — matrix operand order repaired and checked by reading, not compiled, not run on hardware
//
#include <hip/hip_runtime.h>
#include <math.h>

typedef __attribute__((ext_vector_type(16))) __bf16          v16bf;
typedef __attribute__((ext_vector_type(16))) unsigned short  v16u;
typedef __attribute__((ext_vector_type(8)))  unsigned short  v8u;
typedef __attribute__((ext_vector_type(8)))  float           v8f;
typedef __attribute__((ext_vector_type(4)))  float           v4f;
typedef __attribute__((ext_vector_type(2)))  unsigned        v2ui;

#define B_ROWS   512
#define DIM      512
#define NCLS     100000
#define S_SCALEF 64.0f
#define SM_F     (64.0f * 0.35f)
#define EPSN     1e-5f

#define NTILE    64                              // classes per workgroup
#define N_CT     ((NCLS + NTILE - 1) / NTILE)    // 1563 class tiles
#define WROW     40                              // LDS row stride in ushorts (80 B)

// pack two fp32 -> packed bf16 (low = f0), round-half-to-nearest: 3 VALU
__device__ __forceinline__ unsigned pack_bf16_pair(float f0, float f1) {
    unsigned u0 = __builtin_bit_cast(unsigned, f0) + 0x8000u;
    unsigned u1 = __builtin_bit_cast(unsigned, f1) + 0x8000u;
    return __builtin_amdgcn_perm(u1, u0, 0x07060302u);   // {hi16(u1), hi16(u0)}
}

__device__ __forceinline__ unsigned short f2bf(float f) {
    return (unsigned short)((__builtin_bit_cast(unsigned, f) + 0x8000u) >> 16);
}

// ---------------------------------------------------------------- kernel 1
// L2-normalize each row of x, store bf16 (one block per row)
__global__ __launch_bounds__(256) void k_norm_x(const float* __restrict__ x,
                                                unsigned short* __restrict__ xbf) {
    const int row = blockIdx.x;
    const float* xr = x + (size_t)row * DIM;
    float ss = 0.f;
    for (int k = threadIdx.x; k < DIM; k += 256) { float v = xr[k]; ss = fmaf(v, v, ss); }
    for (int m = 16; m >= 1; m >>= 1) ss += __shfl_xor(ss, m, 32);
    __shared__ float wsum[8];
    const int wid = threadIdx.x >> 5;
    if ((threadIdx.x & 31) == 0) wsum[wid] = ss;
    __syncthreads();
    float tot = 0.f;
    #pragma unroll
    for (int i = 0; i < 8; ++i) tot += wsum[i];
    const float r = 1.f / fmaxf(sqrtf(tot), EPSN);
    for (int k = threadIdx.x; k < DIM; k += 256)
        xbf[(size_t)row * DIM + k] = f2bf(xr[k] * r);
}

// ---------------------------------------------------------------- kernel 2
// BF16 WMMA GEMM tile with LDS-staged, once-converted W (double-buffered),
// fused W-row-norm and fused exp-sum.
// grid.x = N_CT class tiles of 64; block = 512 threads (16 waves: 8 M x 2 N)
__global__ __launch_bounds__(512, 1) void k_gemm(const unsigned short* __restrict__ xbf,
                                                 const float* __restrict__ W,
                                                 float* __restrict__ partial) {
    __shared__ unsigned short wtile[2][NTILE][WROW];  // bf16 W tile, 80 B rows
    __shared__ float sqpart[NTILE][8];                // per (class, k-quad) |w|^2
    __shared__ float rowsum[2][B_ROWS];               // per N-wave exp sums

    const int tid   = threadIdx.x;
    const int lane  = tid & 31;
    const int l16   = lane & 15;
    const int half  = (lane >> 4) & 1;
    const int wave  = tid >> 5;                 // 0..15
    const int waveM = wave >> 1;                // 0..7 -> rows 64*waveM
    const int waveN = wave & 1;                 // 0..1 -> classes 32*waveN
    const int rowb  = waveM * 64;
    const int cb    = (int)blockIdx.x * NTILE;  // first class of this tile

    // ---- producer mapping: thread owns (class prow, k-quad pq) of the tile
    const int prow = tid >> 3;                  // 0..63
    const int pq   = (tid & 7) * 4;             // 0,4,...,28
    const int pcls = cb + prow;
    const int pcrd = pcls < NCLS ? pcls : (NCLS - 1);
    const float* wrow = W + (size_t)pcrd * DIM + pq;
    float psq = 0.f;

    // prologue: stage tile for kk=0
    {
        v4f w = *(const v4f*)(wrow);
        psq = fmaf(w[0], w[0], psq); psq = fmaf(w[1], w[1], psq);
        psq = fmaf(w[2], w[2], psq); psq = fmaf(w[3], w[3], psq);
        v2ui p = { pack_bf16_pair(w[0], w[1]), pack_bf16_pair(w[2], w[3]) };
        *(v2ui*)&wtile[0][prow][pq] = p;
    }
    __syncthreads();

    v8f acc[4][2];
    #pragma unroll
    for (int i = 0; i < 4; ++i)
        #pragma unroll
        for (int j = 0; j < 2; ++j)
            acc[i][j] = (v8f)(0.f);

    int buf = 0;
    for (int kk = 0; kk < DIM; kk += 32) {
        const bool more = (kk + 32) < DIM;
        v4f wn;
        if (more) {
            wn = *(const v4f*)(wrow + kk + 32);        // next tile, early issue
            __builtin_prefetch(wrow + kk + 96, 0, 0);  // two tiles ahead
        }

        // --- B fragments from LDS (32x16 bf16): lane = N%16,
        //     elems 0..15 = K half*16+0..15 (contiguous in wtile row)
        v16bf bfr[2];
        #pragma unroll
        for (int j = 0; j < 2; ++j) {
            const unsigned short* bp = &wtile[buf][waveN * 32 + j * 16 + l16][half * 16];
            v8u b0 = *(const v8u*)(bp);
            v8u b1 = *(const v8u*)(bp + 8);
            v16u ub = __builtin_shufflevector(b0, b1,
                        0, 1, 2, 3, 4, 5, 6, 7, 8, 9, 10, 11, 12, 13, 14, 15);
            bfr[j] = __builtin_bit_cast(v16bf, ub);
        }
        // --- A fragments from global bf16 (L2-resident): lane = M%16,
        //     elems 0..15 = K half*16+0..15 (contiguous)
        #pragma unroll
        for (int i = 0; i < 4; ++i) {
            const unsigned short* ap =
                xbf + (size_t)(rowb + i * 16 + l16) * DIM + kk + half * 16;
            v8u a0 = *(const v8u*)(ap);
            v8u a1 = *(const v8u*)(ap + 8);
            v16u ua = __builtin_shufflevector(a0, a1,
                        0, 1, 2, 3, 4, 5, 6, 7, 8, 9, 10, 11, 12, 13, 14, 15);
            v16bf afr = __builtin_bit_cast(v16bf, ua);
            acc[i][0] = __builtin_amdgcn_wmma_f32_16x16x32_bf16(
                false, afr, false, bfr[0], (short)0, acc[i][0], false, false);
            acc[i][1] = __builtin_amdgcn_wmma_f32_16x16x32_bf16(
                false, afr, false, bfr[1], (short)0, acc[i][1], false, false);
        }

        // --- stage next tile into the other buffer (converted exactly once)
        if (more) {
            psq = fmaf(wn[0], wn[0], psq); psq = fmaf(wn[1], wn[1], psq);
            psq = fmaf(wn[2], wn[2], psq); psq = fmaf(wn[3], wn[3], psq);
            v2ui p = { pack_bf16_pair(wn[0], wn[1]), pack_bf16_pair(wn[2], wn[3]) };
            *(v2ui*)&wtile[buf ^ 1][prow][pq] = p;
        }
        __syncthreads();
        buf ^= 1;
    }

    // ---- per-class 1/||W_c||: publish producer partials, sum 8 quads
    sqpart[prow][tid & 7] = psq;
    __syncthreads();

    int cls[2];
    cls[0] = cb + waveN * 32 + l16;
    cls[1] = cb + waveN * 32 + 16 + l16;
    float rn[2], valid[2];
    #pragma unroll
    for (int j = 0; j < 2; ++j) {
        const int cl = waveN * 32 + j * 16 + l16;
        v4f s0 = *(const v4f*)&sqpart[cl][0];
        v4f s1 = *(const v4f*)&sqpart[cl][4];
        float t = ((s0[0] + s0[1]) + (s0[2] + s0[3]))
                + ((s1[0] + s1[1]) + (s1[2] + s1[3]));
        rn[j] = 1.f / fmaxf(sqrtf(t), EPSN);
        valid[j] = (cls[j] < NCLS) ? 1.f : 0.f;
    }

    // ---- exp(64*cos - 64), butterfly-reduce over this wave's 32 classes
    #pragma unroll
    for (int i = 0; i < 4; ++i) {
        float ps[8];
        #pragma unroll
        for (int v = 0; v < 8; ++v) {
            float e = valid[0] * __expf(S_SCALEF * acc[i][0][v] * rn[0] - S_SCALEF)
                    + valid[1] * __expf(S_SCALEF * acc[i][1][v] * rn[1] - S_SCALEF);
            e += __shfl_xor(e, 1, 32);
            e += __shfl_xor(e, 2, 32);
            e += __shfl_xor(e, 4, 32);
            e += __shfl_xor(e, 8, 32);
            ps[v] = e;
        }
        if (l16 == 0) {                 // lanes 0 and 16: rows +0 / +8
            const int row = rowb + i * 16 + half * 8;
            #pragma unroll
            for (int v = 0; v < 8; ++v)
                rowsum[waveN][row + v] = ps[v];   // unique writer per slot
        }
    }
    __syncthreads();
    for (int r = tid; r < B_ROWS; r += 512)
        partial[(size_t)blockIdx.x * B_ROWS + r] = rowsum[0][r] + rowsum[1][r];
}

// ---------------------------------------------------------------- kernel 3
// exact fp32 label logit per row: 64 * cos(x_r, W_label[r])
__global__ __launch_bounds__(256) void k_label(const float* __restrict__ x,
                                               const float* __restrict__ W,
                                               const int* __restrict__ label,
                                               float* __restrict__ lab) {
    const int r = blockIdx.x;
    const int c = label[r];
    const float* xr = x + (size_t)r * DIM;
    const float* wr = W + (size_t)c * DIM;
    float sx = 0.f, sw = 0.f, sd = 0.f;
    for (int k = threadIdx.x; k < DIM; k += 256) {
        float a = xr[k], b = wr[k];
        sx = fmaf(a, a, sx); sw = fmaf(b, b, sw); sd = fmaf(a, b, sd);
    }
    for (int m = 16; m >= 1; m >>= 1) {
        sx += __shfl_xor(sx, m, 32);
        sw += __shfl_xor(sw, m, 32);
        sd += __shfl_xor(sd, m, 32);
    }
    __shared__ float red[3][8];
    const int wid = threadIdx.x >> 5;
    if ((threadIdx.x & 31) == 0) { red[0][wid] = sx; red[1][wid] = sw; red[2][wid] = sd; }
    __syncthreads();
    if (threadIdx.x == 0) {
        float tx = 0.f, tw = 0.f, td = 0.f;
        #pragma unroll
        for (int i = 0; i < 8; ++i) { tx += red[0][i]; tw += red[1][i]; td += red[2][i]; }
        lab[r] = S_SCALEF * td / (fmaxf(sqrtf(tx), EPSN) * fmaxf(sqrtf(tw), EPSN));
    }
}

// ---------------------------------------------------------------- kernel 4
// per-row: fixed-order sum over class tiles, swap in margined label term,
// logsumexp with fixed max = 64, per-row loss
__global__ __launch_bounds__(256) void k_rowloss(const float* __restrict__ partial,
                                                 const float* __restrict__ lab,
                                                 float* __restrict__ rowloss) {
    const int r = blockIdx.x;
    float s = 0.f;
    for (int t = threadIdx.x; t < N_CT; t += 256)
        s += partial[(size_t)t * B_ROWS + r];
    for (int m = 16; m >= 1; m >>= 1) s += __shfl_xor(s, m, 32);
    __shared__ float red[8];
    const int wid = threadIdx.x >> 5;
    if ((threadIdx.x & 31) == 0) red[wid] = s;
    __syncthreads();
    if (threadIdx.x == 0) {
        float tot = 0.f;
        #pragma unroll
        for (int i = 0; i < 8; ++i) tot += red[i];
        const float lp   = lab[r];
        const float lm   = lp - SM_F;
        const float corr = tot - __expf(lp - S_SCALEF) + __expf(lm - S_SCALEF);
        rowloss[r] = S_SCALEF + logf(corr) - lm;
    }
}

// ---------------------------------------------------------------- kernel 5
__global__ __launch_bounds__(512) void k_mean(const float* __restrict__ rowloss,
                                              float* __restrict__ out) {
    float loss = rowloss[threadIdx.x];
    for (int m = 16; m >= 1; m >>= 1) loss += __shfl_xor(loss, m, 32);
    __shared__ float red[16];
    const int wid = threadIdx.x >> 5;
    if ((threadIdx.x & 31) == 0) red[wid] = loss;
    __syncthreads();
    if (threadIdx.x == 0) {
        float tot = 0.f;
        #pragma unroll
        for (int i = 0; i < 16; ++i) tot += red[i];
        out[0] = tot / (float)B_ROWS;
    }
}

// ---------------------------------------------------------------- launch
extern "C" void kernel_launch(void* const* d_in, const int* in_sizes, int n_in,
                              void* d_out, int out_size, void* d_ws, size_t ws_size,
                              hipStream_t stream) {
    const float* x     = (const float*)d_in[0];
    const float* W     = (const float*)d_in[1];
    const int*   label = (const int*)d_in[2];
    float*       out   = (float*)d_out;

    // workspace layout
    unsigned short* xbf = (unsigned short*)d_ws;                        // 512*512*2 B
    float* lab     = (float*)((char*)d_ws + (size_t)B_ROWS * DIM * 2);  // 512 f
    float* rowloss = lab + B_ROWS;                                      // 512 f
    float* partial = rowloss + B_ROWS;                                  // N_CT*512 f

    k_norm_x <<<B_ROWS, 256, 0, stream>>>(x, xbf);
    k_label  <<<B_ROWS, 256, 0, stream>>>(x, W, label, lab);
    k_gemm   <<<N_CT,   512, 0, stream>>>(xbf, W, partial);
    k_rowloss<<<B_ROWS, 256, 0, stream>>>(partial, lab, rowloss);
    k_mean   <<<1,      512, 0, stream>>>(rowloss, out);
}
